// GMM_13967233647339
// MI455X (gfx1250) — matrix operand order repaired and checked by reading, not compiled, not run on hardware
//
#include <hip/hip_runtime.h>
#include <hip/hip_bf16.h>
#include <math.h>

// ---------------------------------------------------------------------------
// GMM negative-log-likelihood, normalized by row max.  out[n,k], N=200000, K=D=32.
//
// nll[n,k] = 0.5*x^T A_k x  -  x . y_k  +  cc_k           (A_k = Sigma_k^-1)
// Cast as ONE f16 WMMA GEMM with reduction dim 1056:
//   chunks t=0..31 : A-frag = x[n,t]*x[n,e]  (vec of outer product, built on the fly)
//                    B      = 0.5*A_k[t][e]
//   chunk  t=32    : A-frag = x[n,e],  B = -y_k[e]
// Accumulate fp32 (V_WMMA_F32_16X16X32_F16); epilogue adds cc_k, row-max, divide.
// B (66KB) staged to LDS via GLOBAL_LOAD_ASYNC_TO_LDS_B128 (ASYNCcnt path).
// ---------------------------------------------------------------------------

typedef __attribute__((ext_vector_type(16))) _Float16 v16h;
typedef __attribute__((ext_vector_type(8)))  float    v8f;
typedef __attribute__((ext_vector_type(2)))  _Float16 h2;

__device__ __forceinline__ h2 pkrtz(float a, float b) {
  return __builtin_bit_cast(h2, __builtin_amdgcn_cvt_pkrtz(a, b));
}

#define KMIX        32
#define DDIM        32
#define KSTRIDE_H   1032                    // halves per mixture row in B (padded: 516 dwords, start banks 4*lane)
#define KSTRIDE_B   (KSTRIDE_H * 2)         // 2064 bytes
#define B_BYTES     (KMIX * KSTRIDE_B)      // 66048 bytes of LDS-resident B
#define WS_CC_OFF   B_BYTES                 // float cc[32]
#define WS_Y_OFF    (B_BYTES + 128)        // _Float16 yneg[32][32] (chunk-32 B frags)

// ------------------------- kernel 1: per-mixture prep -----------------------
// One wave per mixture. Gauss-Jordan inversion of SPD Sigma_k in LDS (lane c
// owns columns c and c+32 of [A|I]); logdet from pivots. Emits f16 B row
// (0.5*Ainv), the -y chunk, and cc_k.
__global__ __launch_bounds__(32) void gmm_prep(
    const float* __restrict__ mus, const float* __restrict__ sigmas,
    const float* __restrict__ phis, unsigned char* __restrict__ ws) {
  __shared__ float M[32][66];     // [A | I], padded row
  __shared__ float stage[32];     // current pivot column
  __shared__ float ys[32];
  const int k = blockIdx.x;
  const int c = threadIdx.x;      // 0..31
  const float* S = sigmas + k * DDIM * DDIM;
  for (int r = 0; r < 32; ++r) {
    M[r][c]      = S[r * 32 + c];
    M[r][32 + c] = (r == c) ? 1.0f : 0.0f;
  }
  __syncthreads();
  float logdet = 0.0f;
  for (int p = 0; p < 32; ++p) {
    stage[c] = M[c][p];
    __syncthreads();
    float piv = stage[p];
    logdet += logf(piv);
    float s1 = M[p][c]      / piv;
    float s2 = M[p][32 + c] / piv;
    #pragma unroll 4
    for (int r = 0; r < 32; ++r) {
      if (r != p) {
        float f = stage[r];
        M[r][c]      -= f * s1;
        M[r][32 + c] -= f * s2;
      }
    }
    M[p][c]      = s1;
    M[p][32 + c] = s2;
    __syncthreads();
  }
  // y = Ainv * mu (lane c computes y_c), m = mu.y
  const float* mu = mus + k * DDIM;
  float yd = 0.0f;
  for (int e = 0; e < 32; ++e) yd += M[c][32 + e] * mu[e];
  ys[c] = yd;
  __syncthreads();
  float m = 0.0f;
  for (int e = 0; e < 32; ++e) m += ys[e] * mu[e];
  const float LOG2PI = 1.8378770664093453f;
  float cc = 0.5f * m + 0.5f * (32.0f * LOG2PI + logdet) - logf(phis[k]);

  _Float16* Bk = (_Float16*)(ws + (size_t)k * KSTRIDE_B);
  #pragma unroll 4
  for (int e = 0; e < 32; ++e)
    Bk[c * 32 + e] = (_Float16)(0.5f * M[c][32 + e]);   // chunk c, slot e
  if (c < (KSTRIDE_H - 1024)) Bk[1024 + c] = (_Float16)0.0f;  // pad
  ((_Float16*)(ws + WS_Y_OFF + (size_t)k * 64))[c] = (_Float16)(-yd);
  if (c == 0) *(float*)(ws + WS_CC_OFF + (size_t)k * 4) = cc;
}

// --------------------------- kernel 2: main GEMM ----------------------------
__global__ __launch_bounds__(256) void gmm_main(
    const float* __restrict__ X, const unsigned char* __restrict__ ws,
    float* __restrict__ out, int N) {
  extern __shared__ __align__(16) unsigned char sB[];   // B_BYTES, LDS offset 0
  const int tid  = threadIdx.x;
  const int lane = tid & 31;
  const int wv   = tid >> 5;
  const int l15  = lane & 15;
  const int hi   = (lane >> 4) & 1;            // half-wave: K/e upper range
  const int rowBase = (blockIdx.x * 8 + wv) * 16;
  const bool active = rowBase < N;

  // Async fill of B into LDS: bypasses VGPRs, tracked by ASYNCcnt.
  // No static __shared__ in this kernel -> dynamic LDS base is offset 0.
  {
    #pragma unroll
    for (int it = 0; it < 17; ++it) {
      const int i = it * 256 + tid;             // uint4 index
      if (i < B_BYTES / 16) {
        const unsigned lds_off = (unsigned)(i * 16);
        const unsigned long long gaddr =
            (unsigned long long)(ws + (size_t)i * 16);
        asm volatile("global_load_async_to_lds_b128 %0, %1, off"
                     :: "v"(lds_off), "v"(gaddr) : "memory");
      }
    }
  }

  float xrow[32];
  union HU { v16h v; h2 h[8]; uint4 q[2]; };
  HU eset, yb0, yb1;

  if (active) {
    const int n = rowBase + l15;               // both halves: same 16 rows
    const float4* xp = (const float4*)(X + (size_t)n * 32);
    __builtin_prefetch(xp, 0, 0);
    #pragma unroll
    for (int i = 0; i < 8; ++i) {
      float4 t = xp[i];
      xrow[4*i+0] = t.x; xrow[4*i+1] = t.y; xrow[4*i+2] = t.z; xrow[4*i+3] = t.w;
    }
    // chunk-32 B fragments (constant per lane, keep in registers)
    const uint4* yp = (const uint4*)(ws + WS_Y_OFF);
    yb0.q[0] = yp[l15 * 4 + hi * 2 + 0];
    yb0.q[1] = yp[l15 * 4 + hi * 2 + 1];
    yb1.q[0] = yp[(16 + l15) * 4 + hi * 2 + 0];
    yb1.q[1] = yp[(16 + l15) * 4 + hi * 2 + 1];
    // f16 x[n,e] as A operand, this kernel's contiguous k order: lanes0-15 e={0..15}, lanes16-31 e={16..31}
    const int ebase = hi ? 16 : 0;
    #pragma unroll
    for (int v = 0; v < 8; ++v) {
      const int e0 = ebase + v * 2;
      eset.h[v] = pkrtz(xrow[e0], xrow[e0 + 1]);
    }
  }
  // All async LDS writes landed, then block-wide barrier.
  asm volatile("s_wait_asynccnt 0" ::: "memory");
  __syncthreads();

  if (active) {
    v8f acc0 = {0,0,0,0,0,0,0,0};
    v8f acc1 = {0,0,0,0,0,0,0,0};
    const int bk0 = l15 * KSTRIDE_B + hi * 32;          // col-tile 0: k = l15
    const int bk1 = (16 + l15) * KSTRIDE_B + hi * 32;   // col-tile 1: k = 16+l15
    #pragma unroll
    for (int t = 0; t < 32; ++t) {
      const h2 xt2 = pkrtz(xrow[t], xrow[t]);
      HU af;
      #pragma unroll
      for (int v = 0; v < 8; ++v) af.h[v] = xt2 * eset.h[v];   // v_pk_mul_f16
      HU b0, b1;
      const uint4* p0 = (const uint4*)(sB + bk0 + t * 64);
      b0.q[0] = p0[0]; b0.q[1] = p0[1];
      const uint4* p1 = (const uint4*)(sB + bk1 + t * 64);
      b1.q[0] = p1[0]; b1.q[1] = p1[1];
      acc0 = __builtin_amdgcn_wmma_f32_16x16x32_f16(false, af.v, false, b0.v,
                                                    (short)0, acc0, false, false);
      acc1 = __builtin_amdgcn_wmma_f32_16x16x32_f16(false, af.v, false, b1.v,
                                                    (short)0, acc1, false, false);
    }
    // chunk 32: A-frag is x itself, B = -y_k
    acc0 = __builtin_amdgcn_wmma_f32_16x16x32_f16(false, eset.v, false, yb0.v,
                                                  (short)0, acc0, false, false);
    acc1 = __builtin_amdgcn_wmma_f32_16x16x32_f16(false, eset.v, false, yb1.v,
                                                  (short)0, acc1, false, false);
    // epilogue: + cc_k, row max over 32 k, divide, store
    const float* ccp = (const float*)(ws + WS_CC_OFF);
    const float cc0 = ccp[l15];
    const float cc1 = ccp[16 + l15];
    #pragma unroll
    for (int r = 0; r < 8; ++r) {
      float a0 = acc0[r] + cc0;
      float a1 = acc1[r] + cc1;
      float mx = fmaxf(a0, a1);
      mx = fmaxf(mx, __shfl_xor(mx, 1, 16));
      mx = fmaxf(mx, __shfl_xor(mx, 2, 16));
      mx = fmaxf(mx, __shfl_xor(mx, 4, 16));
      mx = fmaxf(mx, __shfl_xor(mx, 8, 16));
      const float inv = __builtin_amdgcn_rcpf(mx);
      const int row = rowBase + r + hi * 8;     // VGPR r: M=r (lanes0-15) / M=r+8
      out[(size_t)row * 32 + l15]      = a0 * inv;
      out[(size_t)row * 32 + 16 + l15] = a1 * inv;
    }
  }
}

extern "C" void kernel_launch(void* const* d_in, const int* in_sizes, int n_in,
                              void* d_out, int out_size, void* d_ws, size_t ws_size,
                              hipStream_t stream) {
  (void)n_in; (void)out_size; (void)ws_size;
  const float* X    = (const float*)d_in[0];
  const float* mus  = (const float*)d_in[1];
  const float* sig  = (const float*)d_in[2];
  const float* phis = (const float*)d_in[3];
  unsigned char* ws = (unsigned char*)d_ws;
  const int N = in_sizes[0] / DDIM;

  gmm_prep<<<KMIX, 32, 0, stream>>>(mus, sig, phis, ws);
  const int blocks = (N + 127) / 128;            // 8 waves * 16 rows per block
  gmm_main<<<blocks, 256, B_BYTES, stream>>>(X, ws, (float*)d_out, N);
}
